// AttA3_9534827397203
// MI455X (gfx1250) — matrix operand order repaired and checked by reading, not compiled, not run on hardware
//
#include <hip/hip_runtime.h>
#include <hip/hip_bf16.h>
#include <math.h>

typedef __attribute__((ext_vector_type(16))) _Float16 v16h;
typedef __attribute__((ext_vector_type(8)))  _Float16 v8h;
typedef __attribute__((ext_vector_type(2)))  _Float16 v2h;
typedef __attribute__((ext_vector_type(8)))  float    v8f;

#define LEN   4096
#define EMB   300
#define EMBP  320      // EMB padded to multiple of 32 for WMMA K
#define HID   512
#define HALF  256
#define GATES 1024     // 4*HALF
#define NGATE 2048     // both directions
#define OUT   20

__device__ __forceinline__ float sigmoidf_(float x) {
    return 1.0f / (1.0f + __expf(-x));
}

__device__ __forceinline__ v8f wmma_f16(v16h a, v16h b, v8f c) {
    return __builtin_amdgcn_wmma_f32_16x16x32_f16(false, a, false, b, (short)0, c,
                                                  false, false);
}

// A 16x32 f16 fragment: lane half hf holds K = {8hf..8hf+7} and {16+8hf..16+8hf+7}
__device__ __forceinline__ v16h load_afrag(const _Float16* __restrict__ arow,
                                           int k0, int hf) {
    v8h lo = *(const v8h*)(arow + k0 + hf * 8);
    v8h hi = *(const v8h*)(arow + k0 + 16 + hf * 8);
    v16h a;
#pragma unroll
    for (int i = 0; i < 8; ++i) { a[i] = lo[i]; a[i + 8] = hi[i]; }
    return a;
}

// ---------------------------------------------------------------------------
// Embedding gather -> f16, K padded 300->320
// grid: LEN blocks, 320 threads
__global__ void k_embed(const int* __restrict__ sent, const float* __restrict__ emb,
                        _Float16* __restrict__ x16) {
    const int t = blockIdx.x, k = threadIdx.x;
    const int id = sent[t];
    x16[(size_t)t * EMBP + k] =
        (k < EMB) ? (_Float16)emb[(size_t)id * EMB + k] : (_Float16)0.0f;
}

// ---------------------------------------------------------------------------
// Pack Wih_f/Wih_b -> f16 [2048, 320], and bias_sum[2048] = bih+bhh
// grid: 2048 blocks, 320 threads
__global__ void k_prep_wih(const float* __restrict__ wf, const float* __restrict__ wb,
                           const float* __restrict__ bihf, const float* __restrict__ bhhf,
                           const float* __restrict__ bihb, const float* __restrict__ bhhb,
                           _Float16* __restrict__ w16, float* __restrict__ bsum) {
    const int j = blockIdx.x, k = threadIdx.x;
    const float* src = (j < GATES) ? (wf + (size_t)j * EMB)
                                   : (wb + (size_t)(j - GATES) * EMB);
    w16[(size_t)j * EMBP + k] = (k < EMB) ? (_Float16)src[k] : (_Float16)0.0f;
    if (k == 0)
        bsum[j] = (j < GATES) ? (bihf[j] + bhhf[j]) : (bihb[j - GATES] + bhhb[j - GATES]);
}

// W_h_a [512,512] -> f16
__global__ void k_prep_wha(const float* __restrict__ w, _Float16* __restrict__ w16) {
    const int j = blockIdx.x, k = threadIdx.x;
    w16[(size_t)j * HID + k] = (_Float16)w[(size_t)j * HID + k];
}

// Transpose+pack Whh per direction into f16, K-pairs interleaved:
// wT[dir][k/2][j][k&1] = Whh_dir[j][k]  (one dword per thread = 2 K values)
// grid: dim3(256, 2), 1024 threads
__global__ void k_prep_whhT(const float* __restrict__ whf, const float* __restrict__ whb,
                            _Float16* __restrict__ wT) {
    const int k = blockIdx.x, dir = blockIdx.y, j = threadIdx.x;
    const float* w = dir ? whb : whf;
    wT[(((size_t)dir * (HALF / 2) + (k >> 1)) * GATES + j) * 2 + (k & 1)] =
        (_Float16)w[(size_t)j * HALF + k];
}

// ---------------------------------------------------------------------------
// WMMA GEMM, 2x2 register-blocked: each wave computes a 32x32 output tile
// (4 WMMAs per K-step, reusing 2 A frags + 2 B frags -> ~16 FLOP/byte).
// D[M,N] = A[M,K](f16 rm) x B[N,K](f16 rm)^T + bias[N]
// block = 128 threads = 4 waves along N; grid: dim3(M/32, N/128)
__global__ void k_gemm16(const _Float16* __restrict__ A, const _Float16* __restrict__ B,
                         const float* __restrict__ bias, float* __restrict__ D,
                         int N, int K) {
    const int lane = threadIdx.x & 31;
    const int wave = threadIdx.x >> 5;
    const int ml   = lane & 15;
    const int hf   = lane >> 4;
    const int bm   = blockIdx.x;              // 32-row tile
    const int bn   = blockIdx.y * 4 + wave;   // 32-col tile

    const _Float16* arow0 = A + (size_t)(bm * 32 + ml) * K;
    const _Float16* arow1 = arow0 + (size_t)16 * K;
    const _Float16* brow0 = B + (size_t)(bn * 32 + ml) * K;
    const _Float16* brow1 = brow0 + (size_t)16 * K;

    v8f acc00 = {}, acc01 = {}, acc10 = {}, acc11 = {};
    for (int k0 = 0; k0 < K; k0 += 32) {
        v16h a0 = load_afrag(arow0, k0, hf);
        v16h a1 = load_afrag(arow1, k0, hf);
        // B 32x16 frag: lane half hf holds K = {8hf..8hf+7} and {16+8hf..16+8hf+7}, column n = lane&15
        v16h b0 = load_afrag(brow0, k0, hf);
        v16h b1 = load_afrag(brow1, k0, hf);
        acc00 = wmma_f16(a0, b0, acc00);
        acc01 = wmma_f16(a0, b1, acc01);
        acc10 = wmma_f16(a1, b0, acc10);
        acc11 = wmma_f16(a1, b1, acc11);
    }

    const int n0 = bn * 32 + ml;
    const int n1 = n0 + 16;
    const float bv0 = bias[n0];
    const float bv1 = bias[n1];
#pragma unroll
    for (int r = 0; r < 8; ++r) {
        const int m0 = bm * 32 + r + 8 * hf;      // C/D: VGPR r -> M = r + 8*(lane>=16)
        const int m1 = m0 + 16;
        D[(size_t)m0 * N + n0] = acc00[r] + bv0;
        D[(size_t)m0 * N + n1] = acc01[r] + bv1;
        D[(size_t)m1 * N + n0] = acc10[r] + bv0;
        D[(size_t)m1 * N + n1] = acc11[r] + bv1;
    }
}

// ---------------------------------------------------------------------------
// Sequential bi-LSTM scan. 2 blocks (dir), 1024 threads, h/c/gates in LDS.
// Recurrent weights streamed as packed f16 K-pairs (512 KB/step, L2-resident).
__global__ void k_lstm(const float* __restrict__ pre, const _Float16* __restrict__ whhT,
                       float* __restrict__ out2, _Float16* __restrict__ out2h,
                       float* __restrict__ ht) {
    const int dir = blockIdx.x;
    const int tid = threadIdx.x;           // 0..1023 -> one gate pre-activation
    __shared__ float hs[HALF], cs[HALF], gs[GATES];
    if (tid < HALF) { hs[tid] = 0.0f; cs[tid] = 0.0f; }
    __syncthreads();

    const _Float16* wT  = whhT + (size_t)dir * HALF * GATES;
    const float*    prd = pre + dir * GATES;

    for (int step = 0; step < LEN; ++step) {
        const int t = dir ? (LEN - 1 - step) : step;
        float acc = prd[(size_t)t * NGATE + tid];
        if (step + 1 < LEN) {
            const int tn = dir ? (LEN - 2 - step) : (step + 1);
            __builtin_prefetch(&prd[(size_t)tn * NGATE + tid], 0, 3); // global_prefetch_b8
        }
#pragma unroll 8
        for (int kk = 0; kk < HALF / 2; ++kk) {
            v2h w2 = *(const v2h*)(wT + ((size_t)kk * GATES + tid) * 2); // one b32/lane
            acc = fmaf((float)w2[0], hs[2 * kk],     acc);
            acc = fmaf((float)w2[1], hs[2 * kk + 1], acc);
        }
        gs[tid] = acc;
        __syncthreads();
        if (tid < HALF) {
            const float iv = sigmoidf_(gs[tid]);
            const float fv = sigmoidf_(gs[HALF + tid]);
            const float gv = tanhf(gs[2 * HALF + tid]);
            const float ov = sigmoidf_(gs[3 * HALF + tid]);
            const float cv = fmaf(fv, cs[tid], iv * gv);
            cs[tid] = cv;
            const float hv = ov * tanhf(cv);
            hs[tid] = hv;
            const size_t oi = (size_t)t * HID + dir * HALF + tid;
            out2[oi]  = hv;
            out2h[oi] = (_Float16)hv;
        }
        __syncthreads();
    }
    if (tid < HALF) ht[dir * HALF + tid] = hs[tid];
}

// ---------------------------------------------------------------------------
// Wv_ae[o][h] = dot(W_v_a[h], AE[o]) + b_v_a[h]   grid: 20 x 512
__global__ void k_wvae(const float* __restrict__ AE, const float* __restrict__ Wv,
                       const float* __restrict__ bv, float* __restrict__ wvae) {
    const int o = blockIdx.x, h = threadIdx.x;
    __shared__ float ae[EMB];
    if (h < EMB) ae[h] = AE[(size_t)o * EMB + h];
    __syncthreads();
    const float* wr = Wv + (size_t)h * EMB;
    float p = 0.0f;
    for (int k = 0; k < EMB; ++k) p = fmaf(wr[k], ae[k], p);
    wvae[o * HID + h] = p + bv[h];
}

// scores[o][l] = sum_h tanh(Wh_out[l][h] + Wv_ae[o][h]) * w_a[h] + b_w_a
// one wave per (o,l); grid 20*4096/8 blocks x 256 threads
__global__ void k_scores(const float* __restrict__ whout, const float* __restrict__ wvae,
                         const float* __restrict__ wa, const float* __restrict__ bwa,
                         float* __restrict__ sc) {
    const int lane = threadIdx.x & 31;
    const int idx  = blockIdx.x * 8 + (threadIdx.x >> 5);
    const int o = idx >> 12, l = idx & (LEN - 1);
    const float* wh = whout + (size_t)l * HID;
    const float* wv = wvae + o * HID;
    float acc = 0.0f;
    for (int h = lane; h < HID; h += 32)
        acc = fmaf(tanhf(wh[h] + wv[h]), wa[h], acc);
#pragma unroll
    for (int off = 16; off > 0; off >>= 1) acc += __shfl_xor(acc, off, 32);
    if (lane == 0) sc[idx] = acc + bwa[0];
}

// softmax over L per output class.  grid: 20 blocks x 1024 threads
__global__ void k_softmax(const float* __restrict__ sc, float* __restrict__ w) {
    const int o = blockIdx.x, tid = threadIdx.x;
    __shared__ float red[1024];
    const float* s = sc + (size_t)o * LEN;
    float m = -INFINITY;
    for (int l = tid; l < LEN; l += 1024) m = fmaxf(m, s[l]);
    red[tid] = m; __syncthreads();
    for (int st = 512; st > 0; st >>= 1) {
        if (tid < st) red[tid] = fmaxf(red[tid], red[tid + st]);
        __syncthreads();
    }
    const float mx = red[0]; __syncthreads();
    float sum = 0.0f;
    for (int l = tid; l < LEN; l += 1024) sum += __expf(s[l] - mx);
    red[tid] = sum; __syncthreads();
    for (int st = 512; st > 0; st >>= 1) {
        if (tid < st) red[tid] += red[tid + st];
        __syncthreads();
    }
    const float inv = 1.0f / red[0];
    for (int l = tid; l < LEN; l += 1024)
        w[(size_t)o * LEN + l] = __expf(s[l] - mx) * inv;
}

// r0[o][h] = sum_l weights[o][l] * output2[l][h]   grid: 20 x 512
__global__ void k_r0(const float* __restrict__ w, const float* __restrict__ out2,
                     float* __restrict__ r0) {
    const int o = blockIdx.x, h = threadIdx.x;
    const float* wr = w + (size_t)o * LEN;
    float acc = 0.0f;
#pragma unroll 4
    for (int l = 0; l < LEN; ++l)
        acc = fmaf(wr[l], out2[(size_t)l * HID + h], acc);
    r0[o * HID + h] = acc;
}

// xa[h] = dot(W_x_a[h], ht) + b_x_a[h]   grid: 1 x 512
__global__ void k_xa(const float* __restrict__ Wx, const float* __restrict__ bx,
                     const float* __restrict__ ht, float* __restrict__ xa) {
    const int h = threadIdx.x;
    __shared__ float hts[HID];
    hts[h] = ht[h]; __syncthreads();
    const float* wr = Wx + (size_t)h * HID;
    float p = 0.0f;
#pragma unroll 4
    for (int k = 0; k < HID; ++k) p = fmaf(wr[k], hts[k], p);
    xa[h] = p + bx[h];
}

// decoded[o] = sigmoid( sum_h tanh(dot(W_p_a[h], r0[o]) + b_p_a[h] + xa[h]) * dec_W[o][h] + dec_b[o] )
// grid: 20 x 512
__global__ void k_decode(const float* __restrict__ r0, const float* __restrict__ Wp,
                         const float* __restrict__ bp, const float* __restrict__ xa,
                         const float* __restrict__ decW, const float* __restrict__ decb,
                         float* __restrict__ out) {
    const int o = blockIdx.x, h = threadIdx.x;
    __shared__ float rs[HID];
    __shared__ float red[HID];
    rs[h] = r0[o * HID + h];
    __syncthreads();
    const float* wr = Wp + (size_t)h * HID;
    float p = 0.0f;
#pragma unroll 4
    for (int k = 0; k < HID; ++k) p = fmaf(wr[k], rs[k], p);
    red[h] = tanhf(p + bp[h] + xa[h]) * decW[(size_t)o * HID + h];
    __syncthreads();
    for (int st = 256; st > 0; st >>= 1) {
        if (h < st) red[h] += red[h + st];
        __syncthreads();
    }
    if (h == 0) out[o] = sigmoidf_(red[0] + decb[o]);
}

// ---------------------------------------------------------------------------
extern "C" void kernel_launch(void* const* d_in, const int* in_sizes, int n_in,
                              void* d_out, int out_size, void* d_ws, size_t ws_size,
                              hipStream_t stream) {
    (void)in_sizes; (void)n_in; (void)out_size; (void)ws_size;
    const int*   sentence = (const int*)  d_in[0];
    const float* embed    = (const float*)d_in[1];
    const float* Wih_f    = (const float*)d_in[2];
    const float* Whh_f    = (const float*)d_in[3];
    const float* bih_f    = (const float*)d_in[4];
    const float* bhh_f    = (const float*)d_in[5];
    const float* Wih_b    = (const float*)d_in[6];
    const float* Whh_b    = (const float*)d_in[7];
    const float* bih_b    = (const float*)d_in[8];
    const float* bhh_b    = (const float*)d_in[9];
    const float* AE       = (const float*)d_in[10];
    const float* W_h_a    = (const float*)d_in[11];
    const float* b_h_a    = (const float*)d_in[12];
    const float* W_v_a    = (const float*)d_in[13];
    const float* b_v_a    = (const float*)d_in[14];
    const float* w_a      = (const float*)d_in[15];
    const float* b_w_a    = (const float*)d_in[16];
    const float* W_p_a    = (const float*)d_in[17];
    const float* b_p_a    = (const float*)d_in[18];
    const float* W_x_a    = (const float*)d_in[19];
    const float* b_x_a    = (const float*)d_in[20];
    const float* dec_W    = (const float*)d_in[21];
    const float* dec_b    = (const float*)d_in[22];
    float* outp = (float*)d_out;

    char* ws = (char*)d_ws;
    size_t off = 0;
    auto carve = [&](size_t bytes) -> char* {
        char* p = ws + off;
        off = (off + bytes + 255) & ~(size_t)255;
        return p;
    };
    _Float16* xf16   = (_Float16*)carve((size_t)LEN * EMBP * 2);
    _Float16* wih16  = (_Float16*)carve((size_t)NGATE * EMBP * 2);
    float*    bsum   = (float*)   carve((size_t)NGATE * 4);
    _Float16* whhT   = (_Float16*)carve((size_t)2 * HALF * GATES * 2);
    float*    pre    = (float*)   carve((size_t)LEN * NGATE * 4);
    float*    out2   = (float*)   carve((size_t)LEN * HID * 4);
    _Float16* out2h  = (_Float16*)carve((size_t)LEN * HID * 2);
    float*    ht     = (float*)   carve((size_t)HID * 4);
    _Float16* wha16  = (_Float16*)carve((size_t)HID * HID * 2);
    float*    whout  = (float*)   carve((size_t)LEN * HID * 4);
    float*    wvae   = (float*)   carve((size_t)OUT * HID * 4);
    float*    scores = (float*)   carve((size_t)OUT * LEN * 4);
    float*    wsoft  = (float*)   carve((size_t)OUT * LEN * 4);
    float*    r0     = (float*)   carve((size_t)OUT * HID * 4);
    float*    xa     = (float*)   carve((size_t)HID * 4);

    // Stage 1: gather + weight packing (f16 / transpose)
    k_embed<<<LEN, EMBP, 0, stream>>>(sentence, embed, xf16);
    k_prep_wih<<<NGATE, EMBP, 0, stream>>>(Wih_f, Wih_b, bih_f, bhh_f, bih_b, bhh_b,
                                           wih16, bsum);
    k_prep_wha<<<HID, HID, 0, stream>>>(W_h_a, wha16);
    k_prep_whhT<<<dim3(HALF, 2), GATES, 0, stream>>>(Whh_f, Whh_b, whhT);

    // Stage 2: big WMMA GEMM — pre-activations [4096, 2048]
    k_gemm16<<<dim3(LEN / 32, NGATE / 128), 128, 0, stream>>>(xf16, wih16, bsum, pre,
                                                              NGATE, EMBP);

    // Stage 3: sequential bi-LSTM scan (latency-bound; f16-packed Whh, L2-resident)
    k_lstm<<<2, GATES, 0, stream>>>(pre, whhT, out2, out2h, ht);

    // Stage 4: attention projection WMMA GEMM — Wh_out [4096, 512]
    k_gemm16<<<dim3(LEN / 32, HID / 128), 128, 0, stream>>>(out2h, wha16, b_h_a, whout,
                                                            HID, HID);

    // Stage 5: attention scores / softmax / context
    k_wvae<<<OUT, HID, 0, stream>>>(AE, W_v_a, b_v_a, wvae);
    k_scores<<<(OUT * LEN) / 8, 256, 0, stream>>>(whout, wvae, w_a, b_w_a, scores);
    k_softmax<<<OUT, 1024, 0, stream>>>(scores, wsoft);
    k_r0<<<OUT, HID, 0, stream>>>(wsoft, out2, r0);

    // Stage 6: decode
    k_xa<<<1, HID, 0, stream>>>(W_x_a, b_x_a, ht, xa);
    k_decode<<<OUT, HID, 0, stream>>>(r0, W_p_a, b_p_a, xa, dec_W, dec_b, outp);
}
